// MoFLayer_9414568312947
// MI455X (gfx1250) — matrix operand order repaired and checked by reading, not compiled, not run on hardware
//
#include <hip/hip_runtime.h>

// Problem constants (match reference)
#define NTOK 8192
#define INDIM 1024
#define OUTDIM 1024
#define NEXP 8

// GEMM tiling
#define BM 128
#define BN 128
#define BK 32
#define KSTR 40                 // padded LDS K-stride (elems): 80B rows -> 16B aligned
#define NITER (INDIM / BK)      // 32 K-steps

typedef __bf16 bf16x4  __attribute__((ext_vector_type(4)));
typedef __bf16 bf16x8  __attribute__((ext_vector_type(8)));
typedef __bf16 bf16x16 __attribute__((ext_vector_type(16)));
typedef float  f32x8   __attribute__((ext_vector_type(8)));

// CDNA5 async LDS copy: DMA 16B from global to LDS, tracked by ASYNCcnt.
// VDST operand carries the wave-relative LDS byte offset (ISA §10 VGLOBAL).
__device__ __forceinline__ void async_copy_b128(unsigned lds_off,
                                                unsigned long long gaddr) {
    asm volatile("global_load_async_to_lds_b128 %0, %1, off"
                 :: "v"(lds_off), "v"(gaddr) : "memory");
}

// ---------------------------------------------------------------------------
// x: f32 -> bf16 (straight copy), 4 elements per thread
// ---------------------------------------------------------------------------
__global__ __launch_bounds__(256) void conv_x(const float* __restrict__ src,
                                              __bf16* __restrict__ dst) {
    const size_t i = ((size_t)blockIdx.x * 256 + threadIdx.x) * 4;
    const float4 v = *(const float4*)(src + i);
    bf16x4 o = { (__bf16)v.x, (__bf16)v.y, (__bf16)v.z, (__bf16)v.w };
    *(bf16x4*)(dst + i) = o;
}

// ---------------------------------------------------------------------------
// W[e][k][n] (f32) -> Wt[e][n][k] (bf16), 32x32 tiled LDS transpose.
// Makes every GEMM tile a contiguous byte-copy (async-LDS friendly).
// ---------------------------------------------------------------------------
__global__ __launch_bounds__(256) void conv_wt(const float* __restrict__ W,
                                               __bf16* __restrict__ Wt) {
    __shared__ float tile[32][33];
    const int e  = blockIdx.z;
    const int k0 = blockIdx.x * 32;
    const int n0 = blockIdx.y * 32;
    const int tx = threadIdx.x & 31;
    const int ty = threadIdx.x >> 5;     // 0..7
    const float* Wb = W  + (size_t)e * INDIM * OUTDIM;
    __bf16*      Wo = Wt + (size_t)e * INDIM * OUTDIM;
#pragma unroll
    for (int i = 0; i < 32; i += 8)
        tile[ty + i][tx] = Wb[(size_t)(k0 + ty + i) * OUTDIM + n0 + tx];
    __syncthreads();
#pragma unroll
    for (int i = 0; i < 32; i += 8)
        Wo[(size_t)(n0 + ty + i) * INDIM + k0 + tx] = (__bf16)tile[tx][ty + i];
}

// ---------------------------------------------------------------------------
// Gating: scores -> top-2 -> 2-way softmax; bias-initialize out; build
// per-expert (token, weight) lists for both routing slots. One wave32/token.
// ---------------------------------------------------------------------------
__global__ __launch_bounds__(256) void moe_gate(
    const float* __restrict__ x,   // [NTOK, INDIM]
    const float* __restrict__ Wg,  // [INDIM, NEXP]
    const float* __restrict__ bg,  // [NEXP]
    const float* __restrict__ b,   // [NEXP, OUTDIM]
    float* __restrict__ out,       // [NTOK, OUTDIM]
    int*   __restrict__ cnt,       // [2*NEXP]
    int*   __restrict__ tlist,     // [2*NEXP, NTOK]
    float* __restrict__ wlist)     // [2*NEXP, NTOK]
{
    const int wave = threadIdx.x >> 5;
    const int lane = threadIdx.x & 31;
    const int n = blockIdx.x * 8 + wave;

    const float* xr = x + (size_t)n * INDIM;

    float acc[NEXP];
#pragma unroll
    for (int e = 0; e < NEXP; ++e) acc[e] = 0.f;

    for (int i = lane; i < INDIM; i += 32) {
        const float xv = xr[i];
        const float4 w0 = *(const float4*)(Wg + (size_t)i * NEXP + 0);
        const float4 w1 = *(const float4*)(Wg + (size_t)i * NEXP + 4);
        acc[0] += xv * w0.x; acc[1] += xv * w0.y;
        acc[2] += xv * w0.z; acc[3] += xv * w0.w;
        acc[4] += xv * w1.x; acc[5] += xv * w1.y;
        acc[6] += xv * w1.z; acc[7] += xv * w1.w;
    }

#pragma unroll
    for (int e = 0; e < NEXP; ++e) {
        float v = acc[e];
#pragma unroll
        for (int off = 16; off > 0; off >>= 1) v += __shfl_xor(v, off, 32);
        acc[e] = v + bg[e];
    }

    // top-2, first-index tie-break (matches jax.lax.top_k)
    int e0 = 0; float s0 = acc[0];
#pragma unroll
    for (int e = 1; e < NEXP; ++e) { if (acc[e] > s0) { s0 = acc[e]; e0 = e; } }
    int e1 = -1; float s1 = -3.0e38f;
#pragma unroll
    for (int e = 0; e < NEXP; ++e) {
        if (e != e0 && acc[e] > s1) { s1 = acc[e]; e1 = e; }
    }

    const float z  = __expf(s1 - s0);
    const float g0 = 1.f / (1.f + z);
    const float g1 = z * g0;

    if (lane == 0) {
        int p0 = atomicAdd(&cnt[e0], 1);
        tlist[(size_t)e0 * NTOK + p0] = n;
        wlist[(size_t)e0 * NTOK + p0] = g0;
        int p1 = atomicAdd(&cnt[NEXP + e1], 1);
        tlist[(size_t)(NEXP + e1) * NTOK + p1] = n;
        wlist[(size_t)(NEXP + e1) * NTOK + p1] = g1;
    }

    // out row = weighted bias term (also clears poisoned output)
    const float* b0 = b + (size_t)e0 * OUTDIM;
    const float* b1 = b + (size_t)e1 * OUTDIM;
    float* orow = out + (size_t)n * OUTDIM;
    for (int c = lane; c < OUTDIM; c += 32)
        orow[c] = g0 * b0[c] + g1 * b1[c];
}

// ---------------------------------------------------------------------------
// Gathered expert GEMM (one routing slot per launch):
//   out[tok,:] += w * (x[tok,:] @ W[e])
// 128x128 tile, BK=32, async-LDS double-buffered staging, 8 waves each doing
// a 64x32 sub-tile via v_wmma_f32_16x16x32_bf16.
// ---------------------------------------------------------------------------
__global__ __launch_bounds__(256) void moe_gemm(
    const __bf16* __restrict__ xb,   // [NTOK, INDIM] bf16
    const __bf16* __restrict__ Wt,   // [NEXP, OUTDIM, INDIM] bf16 (transposed)
    float* __restrict__ out,         // [NTOK, OUTDIM]
    const int*   __restrict__ cnt,   // [2*NEXP]
    const int*   __restrict__ tlist, // [2*NEXP, NTOK]
    const float* __restrict__ wlist, // [2*NEXP, NTOK]
    int slot)
{
    __shared__ __bf16 Al[2][BM * KSTR];
    __shared__ __bf16 Bl[2][BN * KSTR];
    __shared__ int   tok_s[BM];
    __shared__ float w_s[BM];

    const int tilesPerE = NTOK / BM;              // 64
    const int e     = blockIdx.x / tilesPerE;
    const int mt    = blockIdx.x % tilesPerE;
    const int lid   = slot * NEXP + e;
    const int count = cnt[lid];
    const int rbase = mt * BM;
    if (rbase >= count) return;
    const int nbase = blockIdx.y * BN;

    const int t = threadIdx.x;
    if (t < BM) {
        const int idx = rbase + t;
        int tk = -1; float ww = 0.f;
        if (idx < count) {
            tk = tlist[(size_t)lid * NTOK + idx];
            ww = wlist[(size_t)lid * NTOK + idx];
        }
        tok_s[t] = tk;
        w_s[t]  = ww;
    }
    __syncthreads();

    const __bf16* Wb = Wt + (size_t)e * INDIM * OUTDIM + (size_t)nbase * INDIM;

    // Issue the async copies for one K-stage into LDS buffer `buf`.
    // Every thread issues exactly 4 b128 ops (2 A + 2 B) so each wave adds
    // exactly 4 to ASYNCcnt per stage -> s_wait_asynccnt thresholds are exact.
    auto stage = [&](int it, int buf) {
        const int kb = it * BK;
        const unsigned aB = (unsigned)(size_t)&Al[buf][0];
        const unsigned bB = (unsigned)(size_t)&Bl[buf][0];
#pragma unroll
        for (int j = 0; j < 2; ++j) {             // A: gathered token rows
            const int q   = j * 256 + t;          // 512 chunks
            const int row = q >> 2;               // 0..127
            const int c8  = (q & 3) * 8;          // K sub-chunk (elements)
            const int tk  = tok_s[row];
            const __bf16* src =
                xb + (size_t)(tk >= 0 ? tk : 0) * INDIM + kb + c8;
            async_copy_b128(aB + (unsigned)(row * KSTR + c8) * 2,
                            (unsigned long long)(size_t)src);
        }
#pragma unroll
        for (int j = 0; j < 2; ++j) {             // B: rows of W^T (contiguous K)
            const int q   = j * 256 + t;
            const int row = q >> 2;
            const int c8  = (q & 3) * 8;
            const __bf16* src = Wb + (size_t)row * INDIM + kb + c8;
            async_copy_b128(bB + (unsigned)(row * KSTR + c8) * 2,
                            (unsigned long long)(size_t)src);
        }
    };

    const int wave  = t >> 5;
    const int lane  = t & 31;
    const int waveM = wave >> 2;          // 0..1 -> 64-row strip
    const int waveN = wave & 3;           // 0..3 -> 32-col strip
    const int lr    = lane & 15;
    const int hiHalf = (lane >= 16) ? 1 : 0;
    const int halfA = hiHalf * 8;         // A fragment K-half offset
    const int halfB = hiHalf * 8;         // B fragment K-half offset

    f32x8 acc[4][2];
#pragma unroll
    for (int mf = 0; mf < 4; ++mf)
#pragma unroll
        for (int nf = 0; nf < 2; ++nf)
            acc[mf][nf] = (f32x8){0.f,0.f,0.f,0.f,0.f,0.f,0.f,0.f};

    stage(0, 0);                                   // prologue: preload tile 0

    for (int it = 0; it < NITER; ++it) {
        const int buf = it & 1;
        if (it + 1 < NITER) {
            stage(it + 1, buf ^ 1);                // prefetch next tile
            // 4 ops of tile it+1 in flight; tile it guaranteed landed:
            asm volatile("s_wait_asynccnt 0x4" ::: "memory");
        } else {
            asm volatile("s_wait_asynccnt 0x0" ::: "memory");
        }
        __syncthreads();                           // publish tile across waves

        const __bf16* Ab = &Al[buf][0];
        const __bf16* Bb = &Bl[buf][0];

        bf16x16 a[4], bfr[2];
#pragma unroll
        for (int mf = 0; mf < 4; ++mf) {
            const __bf16* p = Ab + (waveM * 64 + mf * 16 + lr) * KSTR + halfA;
            bf16x8 lo = *(const bf16x8*)p;         // K [halfA,    halfA+8)
            bf16x8 hi = *(const bf16x8*)(p + 16);  // K [halfA+16, halfA+24)
            a[mf] = __builtin_shufflevector(lo, hi,
                     0,1,2,3,4,5,6,7,8,9,10,11,12,13,14,15);
        }
#pragma unroll
        for (int nf = 0; nf < 2; ++nf) {
            const __bf16* p = Bb + (waveN * 32 + nf * 16 + lr) * KSTR + halfB;
            bf16x8 lo = *(const bf16x8*)p;         // K [halfB,    halfB+8)
            bf16x8 hi = *(const bf16x8*)(p + 16);  // K [halfB+16, halfB+24)
            bfr[nf] = __builtin_shufflevector(lo, hi,
                     0,1,2,3,4,5,6,7,8,9,10,11,12,13,14,15);
        }

#pragma unroll
        for (int mf = 0; mf < 4; ++mf)
#pragma unroll
            for (int nf = 0; nf < 2; ++nf)
                acc[mf][nf] = __builtin_amdgcn_wmma_f32_16x16x32_bf16(
                    false, a[mf], false, bfr[nf],
                    (short)0, acc[mf][nf], false, false);

        __syncthreads();                           // free buffer for reuse
    }

    // Epilogue: scale by gathered gate weight, scatter-accumulate by token id.
    const int rowOff = hiHalf * 8;    // f32 C/D layout: VGPR r -> M = r + 8*hi
#pragma unroll
    for (int mf = 0; mf < 4; ++mf) {
        const int rb = waveM * 64 + mf * 16 + rowOff;
#pragma unroll
        for (int r = 0; r < 8; ++r) {
            const int row = rb + r;
            const int tk  = tok_s[row];
            if (tk < 0) continue;
            const float w = w_s[row];
            float* orow = out + (size_t)tk * OUTDIM;
#pragma unroll
            for (int nf = 0; nf < 2; ++nf) {
                const int col = nbase + waveN * 32 + nf * 16 + lr;
                orow[col] += w * acc[mf][nf][r];
            }
        }
    }
}

// ---------------------------------------------------------------------------
// Launcher
// ---------------------------------------------------------------------------
extern "C" void kernel_launch(void* const* d_in, const int* in_sizes, int n_in,
                              void* d_out, int out_size, void* d_ws, size_t ws_size,
                              hipStream_t stream) {
    const float* x  = (const float*)d_in[0];   // [N, IN]
    const float* W  = (const float*)d_in[1];   // [E, IN, OUT]
    const float* b  = (const float*)d_in[2];   // [E, OUT]
    const float* Wg = (const float*)d_in[3];   // [IN, E]
    const float* bg = (const float*)d_in[4];   // [E]
    float* out = (float*)d_out;                // [N, OUT]

    // workspace layout (~34 MB):
    //   [0,1KB)      cnt (16 ints)
    //   [1KB, ...)   tlist 16*NTOK ints | wlist 16*NTOK floats
    //   [2MB, 18MB)  xb : bf16 x
    //   [18MB,34MB)  Wt : bf16 W transposed [E][OUT][IN]
    char* ws = (char*)d_ws;
    int*    cnt   = (int*)ws;
    int*    tlist = (int*)(ws + 1024);
    float*  wlist = (float*)(ws + 1024 + (size_t)2 * NEXP * NTOK * sizeof(int));
    __bf16* xb    = (__bf16*)(ws + (2u << 20));
    __bf16* Wt    = (__bf16*)(ws + (2u << 20) + (size_t)NTOK * INDIM * sizeof(__bf16));

    hipMemsetAsync(cnt, 0, 2 * NEXP * sizeof(int), stream);

    conv_x<<<(NTOK * INDIM) / 1024, 256, 0, stream>>>(x, xb);
    conv_wt<<<dim3(INDIM / 32, OUTDIM / 32, NEXP), 256, 0, stream>>>(W, Wt);
    moe_gate<<<NTOK / 8, 256, 0, stream>>>(x, Wg, bg, b, out, cnt, tlist, wlist);

    dim3 grid(NEXP * (NTOK / BM), OUTDIM / BN);   // 512 x 8, most tiles early-exit
    // slot 0 then slot 1: stream order serializes the out[] read-modify-write
    moe_gemm<<<grid, 256, 0, stream>>>(xb, Wt, out, cnt, tlist, wlist, 0);
    moe_gemm<<<grid, 256, 0, stream>>>(xb, Wt, out, cnt, tlist, wlist, 1);
}
